// BotGCN_5901285065195
// MI455X (gfx1250) — matrix operand order repaired and checked by reading, not compiled, not run on hardware
//
#include <hip/hip_runtime.h>
#include <hip/hip_bf16.h>

typedef __attribute__((ext_vector_type(16))) __bf16 v16bf;
typedef __attribute__((ext_vector_type(4)))  __bf16 v4bf;
typedef __attribute__((ext_vector_type(8)))  float  v8f;

__device__ __forceinline__ float lrelu(float v) { return v >= 0.f ? v : 0.01f * v; }

// A-fragment LDS slot for (row-in-tile r, k-in-chunk k):
//   lane half h = (k>>3)&1, packed element e = (k&7)|((k&16)>>1)
__device__ __forceinline__ int a_slot(int r, int k) {
    int h = (k >> 3) & 1;
    int e = (k & 7) | ((k & 16) >> 1);
    return (((r >> 4) * 32) + h * 16 + (r & 15)) * 16 + e;
}
// B-fragment LDS slot for (k-in-chunk kk, col n): lane = ((kk>>3)&1)*16 + (n&15), e = (kk&7)|((kk&16)>>1)
__device__ __forceinline__ int b_slot(int kk, int n) {
    return (((n >> 4) * 32) + (((kk >> 3) & 1) << 4) + (n & 15)) * 16 + ((kk & 7) | ((kk & 16) >> 1));
}

// ---------------------------------------------------------------------------
// WMMA GEMM: C[n x Fout] = act( A[n x K] @ W[K x Fout] + bias ), bf16 inputs
// converted on the fly, f32 accumulate via v_wmma_f32_16x16x32_bf16.
// 256 threads = 8 waves; wave w owns rows [row0+16w, row0+16w+16), iterates
// NT 16-col tiles. LDS holds pre-swizzled fragments: one contiguous 32B
// read per lane per operand.
// ---------------------------------------------------------------------------
template<int NT>
__global__ __launch_bounds__(256)
void wmma_gemm_kernel(const float* __restrict__ A, const float* __restrict__ W,
                      const float* __restrict__ bias, float* __restrict__ C,
                      int nRows, int K, int Fout, int ldC, int colOff, int act)
{
    __shared__ __bf16 Af[8 * 32 * 16];      // [rowSlab][lane][elem]
    __shared__ __bf16 Bf[NT * 32 * 16];     // [colTile][lane][elem]

    const int tid  = threadIdx.x;
    const int wave = tid >> 5;
    const int lane = tid & 31;
    const int half = lane >> 4;
    const int row0 = blockIdx.x * 128;

    const bool rowsFull = (row0 + 128) <= nRows;
    const bool colsFull = (NT * 16) == Fout;

    v8f acc[NT] = {};

    for (int k0 = 0; k0 < K; k0 += 32) {
        const bool kFull = (k0 + 32) <= K;

        // ---- stage A tile: 128 rows x 32 K ----
        if (rowsFull & kFull) {
            // 1024 float4s; thread does 4, fully coalesced, no guards.
#pragma unroll
            for (int i = 0; i < 4; ++i) {
                int g = tid + i * 256;
                int r = g >> 3, k = (g & 7) << 2;
                const float4 v = *(const float4*)(A + (size_t)(row0 + r) * K + (k0 + k));
                v4bf p;
                p.x = (__bf16)v.x; p.y = (__bf16)v.y;
                p.z = (__bf16)v.z; p.w = (__bf16)v.w;
                *(v4bf*)&Af[a_slot(r, k)] = p;   // e0 % 4 == 0 -> 8B aligned
            }
        } else {
            // Phase 1: clamped unconditional loads into regs (loads pipeline,
            // no EXEC churn, one wait amortized over all 16).
            float vals[16];
#pragma unroll
            for (int i = 0; i < 16; ++i) {
                int f = tid + i * 256;
                int r = f >> 5, k = f & 31;
                int grc = min(row0 + r, nRows - 1);
                int kc  = min(k0 + k, K - 1);
                vals[i] = A[(size_t)grc * K + kc];
            }
            // Phase 2: predicate-zero + swizzled LDS stores.
#pragma unroll
            for (int i = 0; i < 16; ++i) {
                int f = tid + i * 256;
                int r = f >> 5, k = f & 31;
                bool ok = ((row0 + r) < nRows) && ((k0 + k) < K);
                Af[a_slot(r, k)] = (__bf16)(ok ? vals[i] : 0.f);
            }
        }

        // ---- stage B tile: 32 K x NT*16 cols ----
        if (kFull & colsFull) {
            constexpr int NF4 = NT * 128;    // float4 count
#pragma unroll
            for (int g = tid; g < NF4; g += 256) {
                int kk = g / (NT * 4);
                int n0 = (g % (NT * 4)) << 2;
                const float4 v = *(const float4*)(W + (size_t)(k0 + kk) * Fout + n0);
                Bf[b_slot(kk, n0 + 0)] = (__bf16)v.x;
                Bf[b_slot(kk, n0 + 1)] = (__bf16)v.y;
                Bf[b_slot(kk, n0 + 2)] = (__bf16)v.z;
                Bf[b_slot(kk, n0 + 3)] = (__bf16)v.w;
            }
        } else {
            constexpr int NB  = 32 * NT * 16;
            constexpr int NIT = (NB + 255) / 256;
            float vals[NIT];
#pragma unroll
            for (int i = 0; i < NIT; ++i) {
                int f = tid + i * 256;
                if (NB % 256 == 0 || f < NB) {
                    int kk  = f / (NT * 16), n = f % (NT * 16);
                    int kkc = min(k0 + kk, K - 1);
                    int nc  = min(n, Fout - 1);
                    vals[i] = W[(size_t)kkc * Fout + nc];
                }
            }
#pragma unroll
            for (int i = 0; i < NIT; ++i) {
                int f = tid + i * 256;
                if (NB % 256 == 0 || f < NB) {
                    int kk = f / (NT * 16), n = f % (NT * 16);
                    bool ok = ((k0 + kk) < K) && (n < Fout);
                    Bf[b_slot(kk, n)] = (__bf16)(ok ? vals[i] : 0.f);
                }
            }
        }
        __syncthreads();

        v16bf a = *(const v16bf*)&Af[(wave * 32 + lane) * 16];
#pragma unroll
        for (int j = 0; j < NT; ++j) {
            v16bf b = *(const v16bf*)&Bf[(j * 32 + lane) * 16];
            acc[j] = __builtin_amdgcn_wmma_f32_16x16x32_bf16(
                false, a, false, b, (short)0, acc[j], false, false);
        }
        __syncthreads();
    }

    // Epilogue: C/D f32 layout -> VGPR r holds M = r + 8*half, N = lane%16.
    const int nLane = lane & 15;
#pragma unroll
    for (int j = 0; j < NT; ++j) {
#pragma unroll
        for (int r = 0; r < 8; ++r) {
            int m   = row0 + wave * 16 + r + half * 8;
            int col = j * 16 + nLane;
            if (m < nRows && col < Fout) {
                float v = acc[j][r];
                if (bias) v += bias[col];
                if (act)  v = lrelu(v);
                C[(size_t)m * ldC + colOff + col] = v;
            }
        }
    }
}

// ---------------------------------------------------------------------------
// Graph aggregation helpers
// ---------------------------------------------------------------------------
__global__ void deg_init_kernel(float* __restrict__ deg, int n) {
    int i = blockIdx.x * blockDim.x + threadIdx.x;
    if (i < n) deg[i] = 1.0f;                        // self-loop contribution
}

__global__ void deg_accum_kernel(const int* __restrict__ dst, float* __restrict__ deg, int E) {
    int e = blockIdx.x * blockDim.x + threadIdx.x;
    if (e < E) unsafeAtomicAdd(&deg[dst[e]], 1.0f);
}

__global__ void dinv_kernel(const float* __restrict__ deg, float* __restrict__ dinv, int n) {
    int i = blockIdx.x * blockDim.x + threadIdx.x;
    if (i < n) { float d = deg[i]; dinv[i] = d > 0.f ? rsqrtf(d) : 0.f; }
}

__global__ void fill_bias_kernel(float* __restrict__ out, const float* __restrict__ bias,
                                 int total, int Dmask) {
    int i = blockIdx.x * blockDim.x + threadIdx.x;
    if (i < total) out[i] = bias[i & Dmask];
}

// One (edge, float4-chunk) pair per thread. Edges [0,E) come from edge_index;
// [E, E+n) are self-loops. 51MB node table is L2-resident on MI455X.
__global__ void scatter_kernel(const float* __restrict__ h, const int* __restrict__ src,
                               const int* __restrict__ dst, const float* __restrict__ dinv,
                               float* __restrict__ out, int E, int n, int cshift) {
    const int chunks = 1 << cshift;
    long long idx   = (long long)blockIdx.x * blockDim.x + threadIdx.x;
    long long total = ((long long)(E + n)) << cshift;
    if (idx >= total) return;
    int edge = (int)(idx >> cshift);
    int c    = (int)(idx & (chunks - 1));
    int s, d;
    if (edge < E) { s = src[edge]; d = dst[edge]; }
    else          { s = edge - E;  d = s; }
    float nrm = dinv[s] * dinv[d];
    const float4 hv = ((const float4*)h)[(size_t)s * chunks + c];
    float* o = out + ((size_t)d << (cshift + 2)) + (c << 2);
    unsafeAtomicAdd(o + 0, hv.x * nrm);
    unsafeAtomicAdd(o + 1, hv.y * nrm);
    unsafeAtomicAdd(o + 2, hv.z * nrm);
    unsafeAtomicAdd(o + 3, hv.w * nrm);
}

// ---------------------------------------------------------------------------
// Host-side launch
// ---------------------------------------------------------------------------
static inline void launch_gemm(const float* A, const float* W, const float* bias, float* C,
                               int n, int K, int Fout, int ldC, int off, int act,
                               hipStream_t s) {
    dim3 g((unsigned)((n + 127) / 128)), b(256);
    int nt = (Fout + 15) / 16;
    if (nt <= 1)
        wmma_gemm_kernel<1><<<g, b, 0, s>>>(A, W, bias, C, n, K, Fout, ldC, off, act);
    else if (nt == 2)
        wmma_gemm_kernel<2><<<g, b, 0, s>>>(A, W, bias, C, n, K, Fout, ldC, off, act);
    else if (nt <= 4)
        wmma_gemm_kernel<4><<<g, b, 0, s>>>(A, W, bias, C, n, K, Fout, ldC, off, act);
    else
        wmma_gemm_kernel<8><<<g, b, 0, s>>>(A, W, bias, C, n, K, Fout, ldC, off, act);
}

extern "C" void kernel_launch(void* const* d_in, const int* in_sizes, int n_in,
                              void* d_out, int out_size, void* d_ws, size_t ws_size,
                              hipStream_t stream) {
    const float* des    = (const float*)d_in[0];
    const float* tweet  = (const float*)d_in[1];
    const float* nump   = (const float*)d_in[2];
    const float* catp   = (const float*)d_in[3];
    const int*   ei     = (const int*)d_in[4];
    const float* W_des  = (const float*)d_in[5];   const float* b_des = (const float*)d_in[6];
    const float* W_twt  = (const float*)d_in[7];   const float* b_twt = (const float*)d_in[8];
    const float* W_num  = (const float*)d_in[9];   const float* b_num = (const float*)d_in[10];
    const float* W_cat  = (const float*)d_in[11];  const float* b_cat = (const float*)d_in[12];
    const float* W_inp  = (const float*)d_in[13];  const float* b_inp = (const float*)d_in[14];
    const float* W_g1   = (const float*)d_in[15];  const float* b_g1  = (const float*)d_in[16];
    const float* W_g2   = (const float*)d_in[17];  const float* b_g2  = (const float*)d_in[18];
    const float* W_o1   = (const float*)d_in[19];  const float* b_o1  = (const float*)d_in[20];
    const float* W_g3   = (const float*)d_in[21];  const float* b_g3  = (const float*)d_in[22];
    const float* W_o2   = (const float*)d_in[23];  const float* b_o2  = (const float*)d_in[24];
    const float* W_o3   = (const float*)d_in[25];  const float* b_o3  = (const float*)d_in[26];
    const float* W_o4   = (const float*)d_in[27];  const float* b_o4  = (const float*)d_in[28];
    const float* W_fin  = (const float*)d_in[29];  const float* b_fin = (const float*)d_in[30];

    const int N = in_sizes[0] / 768;
    const int E = in_sizes[4] / 2;
    const int* src = ei;
    const int* dst = ei + E;

    float* bufA = (float*)d_ws;                 // N*128
    float* bufB = bufA + (size_t)N * 128;       // N*128
    float* bufC = bufB + (size_t)N * 128;       // N*128 (GCN linear output)
    float* deg  = bufC + (size_t)N * 128;       // N
    float* dinv = deg + N;                      // N

    const int T = 256;
    auto blocks = [&](long long w) { return (unsigned)((w + T - 1) / T); };

    // --- symmetric GCN normalization ---
    deg_init_kernel <<<blocks(N), T, 0, stream>>>(deg, N);
    deg_accum_kernel<<<blocks(E), T, 0, stream>>>(dst, deg, E);
    dinv_kernel     <<<blocks(N), T, 0, stream>>>(deg, dinv, N);

    // --- feature encoders -> concat into bufA[N,128] ---
    launch_gemm(des,   W_des, b_des, bufA, N, 768, 32, 128,  0, 1, stream);
    launch_gemm(tweet, W_twt, b_twt, bufA, N, 768, 32, 128, 32, 1, stream);
    launch_gemm(nump,  W_num, b_num, bufA, N,   5, 32, 128, 64, 1, stream);
    launch_gemm(catp,  W_cat, b_cat, bufA, N,   3, 32, 128, 96, 1, stream);

    // --- input projection ---
    launch_gemm(bufA, W_inp, b_inp, bufB, N, 128, 128, 128, 0, 1, stream);

    // --- GCN layer 1 (128 -> 128) ---
    launch_gemm(bufB, W_g1, nullptr, bufC, N, 128, 128, 128, 0, 0, stream);
    fill_bias_kernel<<<blocks((long long)N * 128), T, 0, stream>>>(bufA, b_g1, N * 128, 127);
    scatter_kernel<<<blocks(((long long)(E + N)) << 5), T, 0, stream>>>(bufC, src, dst, dinv, bufA, E, N, 5);

    // --- GCN layer 2 (128 -> 128) ---
    launch_gemm(bufA, W_g2, nullptr, bufC, N, 128, 128, 128, 0, 0, stream);
    fill_bias_kernel<<<blocks((long long)N * 128), T, 0, stream>>>(bufB, b_g2, N * 128, 127);
    scatter_kernel<<<blocks(((long long)(E + N)) << 5), T, 0, stream>>>(bufC, src, dst, dinv, bufB, E, N, 5);

    // --- o1: 128 -> 64, lrelu ---
    launch_gemm(bufB, W_o1, b_o1, bufA, N, 128, 64, 64, 0, 1, stream);

    // --- GCN layer 3 (64 -> 64) ---
    launch_gemm(bufA, W_g3, nullptr, bufC, N, 64, 64, 64, 0, 0, stream);
    fill_bias_kernel<<<blocks((long long)N * 64), T, 0, stream>>>(bufB, b_g3, N * 64, 63);
    scatter_kernel<<<blocks(((long long)(E + N)) << 4), T, 0, stream>>>(bufC, src, dst, dinv, bufB, E, N, 4);

    // --- tail MLP ---
    launch_gemm(bufB, W_o2,  b_o2,  bufA, N, 64, 64, 64, 0, 1, stream);
    launch_gemm(bufA, W_o3,  b_o3,  bufB, N, 64, 32, 32, 0, 1, stream);
    launch_gemm(bufB, W_o4,  b_o4,  bufA, N, 32, 16, 16, 0, 1, stream);
    launch_gemm(bufA, W_fin, b_fin, (float*)d_out, N, 16, 2, 2, 0, 0, stream);
}
